// DoubleVQModel_90340342104363
// MI455X (gfx1250) — compile-verified, matrix operand order checked by reading, not run on hardware
//
#include <hip/hip_runtime.h>
#include <hip/hip_bf16.h>

#define B_   4
#define DIM_ 2048
#define T_   4096
#define F_   256
#define L_   10
#define S_   4
#define K1_  192
#define K2_  48

typedef __attribute__((ext_vector_type(16))) __bf16       v16bf;
typedef __attribute__((ext_vector_type(8)))  __bf16       v8bf;
typedef __attribute__((ext_vector_type(8)))  float        v8f;
typedef __attribute__((ext_vector_type(4)))  unsigned int v4u;
typedef int v4i_ __attribute__((vector_size(16)));   // matches builtin param type

// Async global->LDS path (CDNA5): use if the builtins exist; else reg-staged pipeline.
#if defined(__gfx1250__) && __has_builtin(__builtin_amdgcn_global_load_async_to_lds_b128) && __has_builtin(__builtin_amdgcn_s_wait_asynccnt)
#define USE_ASYNC 1
#else
#define USE_ASYNC 0
#endif

#if USE_ASYNC
// Signature (from probe diagnostics): (int4 AS1* gsrc, int4 AS3* ldst, imm off, imm cpol)
__device__ __forceinline__ void async_cp16(void* lds_dst, const void* gsrc) {
  __builtin_amdgcn_global_load_async_to_lds_b128(
      (__attribute__((address_space(1))) v4i_*)gsrc,
      (__attribute__((address_space(3))) v4i_*)lds_dst, 0, 0);
}
#endif

// GEMM tiling: block = 256 threads = 8 waves (wave32).
// Tile: M=64, N=256, K-step=32. Waves in 2(M) x 4(N) grid,
// each wave owns a 32x64 patch = 2x4 WMMA 16x16 fragments.
// Double-buffered LDS: 2 x (5120 + 20480) = 51.2 KB per block.
#define TM  64
#define TN  256
#define TK  32
#define LDH 40   // LDS row stride in halves: 80B (16B aligned, bank-spread)

// Build a 16x32 bf16 WMMA operand fragment from a padded LDS tile row.
// Layout per ISA 7.12.2 (16-bit A 16x32): lane L<16 holds K {0..7,16..23},
// lane L>=16 holds K {8..15,24..31} of row (L&15).
__device__ __forceinline__ v16bf frag_ld(const __bf16* rowbase, int kh) {
  v8bf lo = *(const v8bf*)(rowbase + kh * 8);
  v8bf hi = *(const v8bf*)(rowbase + 16 + kh * 8);
  return __builtin_shufflevector(lo, hi, 0,1,2,3,4,5,6,7,8,9,10,11,12,13,14,15);
}

// Pointwise GEMM: out[b,t,o] = (act(W[o,:] . X[b,t,:] + bias[o]) [+ res]) [* mask]
// W: bf16 [Cout][Cin]  X: bf16 [B][T][Cin]  res/out: bf16 [B][T][Cout]
// F32OUT: out is f32 [B][Cout][T] (final 2048-ch projection).
template<bool RELU, bool RES, bool F32OUT>
__global__ __launch_bounds__(256)
void gemm_pw(const __bf16* __restrict__ W, const __bf16* __restrict__ X,
             const float* __restrict__ bias, const __bf16* res,
             void* outp, const float* __restrict__ mk, int Cin, int Cout) {
  __shared__ __bf16 As[2][TM * LDH];
  __shared__ __bf16 Bs[2][TN * LDH];
  const int tid  = threadIdx.x;
  const int lane = tid & 31, wave = tid >> 5;
  const int wm = wave >> 2, wn = wave & 3;
  const int bz = blockIdx.z;
  const int n0 = blockIdx.x * TN;
  const int m0 = blockIdx.y * TM;
  const int l15 = lane & 15, kh = lane >> 4;
  const int rowA = tid >> 2, chA = tid & 3;

  v8f acc[2][4] = {};
  const size_t xbase = ((size_t)bz * T_ + n0) * (size_t)Cin;
  const __bf16* wrowA = W + (size_t)(m0 + rowA) * Cin + chA * 8;
  const int NK = Cin / TK;

#if USE_ASYNC
  auto issue = [&](int k0, int buf) {
    async_cp16(&As[buf][rowA * LDH + chA * 8], wrowA + k0);
    #pragma unroll
    for (int j = 0; j < 4; ++j) {
      int task = tid + j * 256;
      int n = task >> 2, ch = task & 3;
      async_cp16(&Bs[buf][n * LDH + ch * 8],
                 &X[xbase + (size_t)n * Cin + k0 + ch * 8]);
    }
  };
  issue(0, 0);
#else
  v4u ra, rb[4];
  auto gload = [&](int k0) {
    ra = *(const v4u*)(wrowA + k0);
    if (k0 + TK < Cin) __builtin_prefetch(wrowA + k0 + TK, 0, 0);
    #pragma unroll
    for (int j = 0; j < 4; ++j) {
      int task = tid + j * 256;
      int n = task >> 2, ch = task & 3;
      rb[j] = *(const v4u*)&X[xbase + (size_t)n * Cin + k0 + ch * 8];
    }
  };
  auto lstore = [&](int buf) {
    *(v4u*)&As[buf][rowA * LDH + chA * 8] = ra;
    #pragma unroll
    for (int j = 0; j < 4; ++j) {
      int task = tid + j * 256;
      int n = task >> 2, ch = task & 3;
      *(v4u*)&Bs[buf][n * LDH + ch * 8] = rb[j];
    }
  };
  gload(0); lstore(0);
#endif

  for (int k = 0; k < NK; ++k) {
    const int cur = k & 1;
#if USE_ASYNC
    if (k + 1 < NK) {
      issue((k + 1) * TK, cur ^ 1);
      __builtin_amdgcn_s_wait_asynccnt(5);  // retire tile k (5 per-wave copies)
    } else {
      __builtin_amdgcn_s_wait_asynccnt(0);
    }
#else
    if (k + 1 < NK) gload((k + 1) * TK);    // overlap global latency w/ compute
#endif
    __syncthreads();

    v16bf af[2], bfv[4];
    #pragma unroll
    for (int mf = 0; mf < 2; ++mf)
      af[mf] = frag_ld(&As[cur][(wm * 32 + mf * 16 + l15) * LDH], kh);
    #pragma unroll
    for (int nf = 0; nf < 4; ++nf)
      bfv[nf] = frag_ld(&Bs[cur][(wn * 64 + nf * 16 + l15) * LDH], kh);
    #pragma unroll
    for (int mf = 0; mf < 2; ++mf)
      #pragma unroll
      for (int nf = 0; nf < 4; ++nf)
        acc[mf][nf] = __builtin_amdgcn_wmma_f32_16x16x32_bf16(
            false, af[mf], false, bfv[nf], (short)0, acc[mf][nf], false, false);
    __syncthreads();
#if !USE_ASYNC
    if (k + 1 < NK) lstore(cur ^ 1);
#endif
  }

  // Epilogue. D layout: lane<16 -> M=r, lane>=16 -> M=8+r; N = lane&15.
  #pragma unroll
  for (int mf = 0; mf < 2; ++mf) {
    #pragma unroll
    for (int nf = 0; nf < 4; ++nf) {
      const int t = n0 + wn * 64 + nf * 16 + l15;
      const int ob = m0 + wm * 32 + mf * 16 + kh * 8;
      const float mval = mk ? mk[(size_t)bz * T_ + t] : 1.0f;
      if constexpr (F32OUT) {
        float* out = (float*)outp;
        #pragma unroll
        for (int r = 0; r < 8; ++r) {
          float v = acc[mf][nf][r] + bias[ob + r];
          if (RELU) v = fmaxf(v, 0.0f);
          out[((size_t)bz * Cout + (ob + r)) * T_ + t] = v * mval;
        }
      } else {
        __bf16* out = (__bf16*)outp;
        const size_t oidx = ((size_t)bz * T_ + t) * (size_t)Cout + ob;
        v8bf rv = {};
        if (RES) rv = *(const v8bf*)&res[oidx];
        v8bf ov;
        #pragma unroll
        for (int r = 0; r < 8; ++r) {
          float v = acc[mf][nf][r] + bias[ob + r];
          if (RELU) v = fmaxf(v, 0.0f);
          if (RES) v += (float)rv[r];
          ov[r] = (__bf16)(v * mval);
        }
        *(v8bf*)&out[oidx] = ov;
      }
    }
  }
}

// Dilated conv (kernel=3) as 3 shifted GEMMs accumulated in WMMA accumulators.
// W: bf16 [3][F][F] (tap-major), X,Y: bf16 [B][T][F]. y = relu(conv + bias).
// Flattened pipeline over 24 (tap,k0) steps, double-buffered like gemm_pw.
__global__ __launch_bounds__(256)
void conv_gemm(const __bf16* __restrict__ W, const __bf16* __restrict__ X,
               const float* __restrict__ bias, __bf16* __restrict__ Y, int dil) {
  __shared__ __bf16 As[2][TM * LDH];
  __shared__ __bf16 Bs[2][TN * LDH];
  const int tid  = threadIdx.x;
  const int lane = tid & 31, wave = tid >> 5;
  const int wm = wave >> 2, wn = wave & 3;
  const int bz = blockIdx.z;
  const int n0 = blockIdx.x * TN;
  const int m0 = blockIdx.y * TM;
  const int l15 = lane & 15, kh = lane >> 4;
  const int rowA = tid >> 2, chA = tid & 3;
  const int NS = 3 * (F_ / TK);  // 24 steps

  v8f acc[2][4] = {};

#if USE_ASYNC
  __shared__ v4u dumpbuf[64];  // per-lane dump slots for OOB async lanes
  auto issueC = [&](int s, int buf) {
    const int tap = s >> 3, k0 = (s & 7) * TK;
    const int shift = (tap - 1) * dil;
    const __bf16* Wt = W + (size_t)tap * F_ * F_;
    async_cp16(&As[buf][rowA * LDH + chA * 8],
               Wt + (size_t)(m0 + rowA) * F_ + k0 + chA * 8);
    #pragma unroll
    for (int j = 0; j < 4; ++j) {
      int task = tid + j * 256;
      int n = task >> 2, ch = task & 3;
      int tsrc = n0 + n + shift;
      bool valid = (tsrc >= 0) && (tsrc < T_);
      if (!valid) {                 // zero-fill padded rows
        v4u z = {};
        *(v4u*)&Bs[buf][n * LDH + ch * 8] = z;
      }
      // Wave-uniform async issue (keeps ASYNCcnt bookkeeping exact under
      // divergence): invalid lanes copy a clamped row into a dump slot.
      void* ldst = valid ? (void*)&Bs[buf][n * LDH + ch * 8]
                         : (void*)&dumpbuf[tid & 63];
      int tclamp = valid ? tsrc : 0;
      async_cp16(ldst, &X[((size_t)bz * T_ + tclamp) * F_ + k0 + ch * 8]);
    }
  };
  issueC(0, 0);
#else
  v4u ra, rb[4];
  auto gloadC = [&](int s) {
    const int tap = s >> 3, k0 = (s & 7) * TK;
    const int shift = (tap - 1) * dil;
    const __bf16* Wt = W + (size_t)tap * F_ * F_;
    ra = *(const v4u*)(Wt + (size_t)(m0 + rowA) * F_ + k0 + chA * 8);
    #pragma unroll
    for (int j = 0; j < 4; ++j) {
      int task = tid + j * 256;
      int n = task >> 2, ch = task & 3;
      int tsrc = n0 + n + shift;
      v4u v = {};
      if (tsrc >= 0 && tsrc < T_)
        v = *(const v4u*)&X[((size_t)bz * T_ + tsrc) * F_ + k0 + ch * 8];
      rb[j] = v;
    }
  };
  auto lstoreC = [&](int buf) {
    *(v4u*)&As[buf][rowA * LDH + chA * 8] = ra;
    #pragma unroll
    for (int j = 0; j < 4; ++j) {
      int task = tid + j * 256;
      int n = task >> 2, ch = task & 3;
      *(v4u*)&Bs[buf][n * LDH + ch * 8] = rb[j];
    }
  };
  gloadC(0); lstoreC(0);
#endif

  for (int s = 0; s < NS; ++s) {
    const int cur = s & 1;
#if USE_ASYNC
    if (s + 1 < NS) {
      issueC(s + 1, cur ^ 1);
      __builtin_amdgcn_s_wait_asynccnt(5);
    } else {
      __builtin_amdgcn_s_wait_asynccnt(0);
    }
#else
    if (s + 1 < NS) gloadC(s + 1);
#endif
    __syncthreads();

    v16bf af[2], bfv[4];
    #pragma unroll
    for (int mf = 0; mf < 2; ++mf)
      af[mf] = frag_ld(&As[cur][(wm * 32 + mf * 16 + l15) * LDH], kh);
    #pragma unroll
    for (int nf = 0; nf < 4; ++nf)
      bfv[nf] = frag_ld(&Bs[cur][(wn * 64 + nf * 16 + l15) * LDH], kh);
    #pragma unroll
    for (int mf = 0; mf < 2; ++mf)
      #pragma unroll
      for (int nf = 0; nf < 4; ++nf)
        acc[mf][nf] = __builtin_amdgcn_wmma_f32_16x16x32_bf16(
            false, af[mf], false, bfv[nf], (short)0, acc[mf][nf], false, false);
    __syncthreads();
#if !USE_ASYNC
    if (s + 1 < NS) lstoreC(cur ^ 1);
#endif
  }

  #pragma unroll
  for (int mf = 0; mf < 2; ++mf) {
    #pragma unroll
    for (int nf = 0; nf < 4; ++nf) {
      const int t = n0 + wn * 64 + nf * 16 + l15;
      const int ob = m0 + wm * 32 + mf * 16 + kh * 8;
      const size_t oidx = ((size_t)bz * T_ + t) * F_ + ob;
      v8bf ov;
      #pragma unroll
      for (int r = 0; r < 8; ++r)
        ov[r] = (__bf16)fmaxf(acc[mf][nf][r] + bias[ob + r], 0.0f);
      *(v8bf*)&Y[oidx] = ov;
    }
  }
}

// x f32 [b][C][T] -> bf16 [b][T][C]
__global__ void transpose_f32_to_bf16(const float* __restrict__ src,
                                      __bf16* __restrict__ dst, int C, int Tlen) {
  __shared__ float tile[32][33];
  const int b = blockIdx.z, c0 = blockIdx.y * 32, t0 = blockIdx.x * 32;
  #pragma unroll
  for (int i = 0; i < 4; ++i)
    tile[threadIdx.y + i * 8][threadIdx.x] =
        src[((size_t)b * C + c0 + threadIdx.y + i * 8) * Tlen + t0 + threadIdx.x];
  __syncthreads();
  #pragma unroll
  for (int i = 0; i < 4; ++i)
    dst[((size_t)b * Tlen + t0 + threadIdx.y + i * 8) * C + c0 + threadIdx.x] =
        (__bf16)tile[threadIdx.x][threadIdx.y + i * 8];
}

// act bf16 [b][t][C] -> f32 [b][C][T] (output_all slices)
__global__ void tc_to_ct_f32(const __bf16* __restrict__ act,
                             float* __restrict__ dst, int C, int Tlen) {
  __shared__ float tile[32][33];
  const int b = blockIdx.z, c0 = blockIdx.y * 32, t0 = blockIdx.x * 32;
  #pragma unroll
  for (int i = 0; i < 4; ++i)
    tile[threadIdx.y + i * 8][threadIdx.x] =
        (float)act[((size_t)b * Tlen + t0 + threadIdx.y + i * 8) * C + c0 + threadIdx.x];
  __syncthreads();
  #pragma unroll
  for (int i = 0; i < 4; ++i)
    dst[((size_t)b * C + c0 + threadIdx.y + i * 8) * Tlen + t0 + threadIdx.x] =
        tile[threadIdx.x][threadIdx.y + i * 8];
}

__global__ void f32_to_bf16_k(const float* __restrict__ s, __bf16* __restrict__ d, size_t n) {
  for (size_t i = (size_t)blockIdx.x * blockDim.x + threadIdx.x; i < n;
       i += (size_t)gridDim.x * blockDim.x)
    d[i] = (__bf16)s[i];
}

// conv weights (l,o,i,k) f32 -> (l,k,o,i) bf16
__global__ void conv_w_reorder(const float* __restrict__ s, __bf16* __restrict__ d, int Ltot) {
  const size_t n = (size_t)Ltot * F_ * F_ * 3;
  for (size_t i = (size_t)blockIdx.x * blockDim.x + threadIdx.x; i < n;
       i += (size_t)gridDim.x * blockDim.x) {
    int k = (int)(i % 3);
    size_t r = i / 3;
    int ic = (int)(r & (F_ - 1)); r >>= 8;
    int o  = (int)(r & (F_ - 1)); r >>= 8;
    int l  = (int)r;
    d[(((size_t)l * 3 + k) * F_ + o) * F_ + ic] = (__bf16)s[i];
  }
}

__global__ void norm_cb(const float* __restrict__ cb, float* __restrict__ en) {
  __shared__ float red[256];
  const int k = blockIdx.x, c = threadIdx.x;
  const float v = cb[(size_t)k * F_ + c];
  red[c] = v * v;
  __syncthreads();
  for (int s = 128; s > 0; s >>= 1) {
    if (c < s) red[c] += red[c + s];
    __syncthreads();
  }
  en[(size_t)k * F_ + c] = v / fmaxf(sqrtf(red[0]), 1e-12f);
}

// wave-per-timestep VQ: cosine argmax, quantize, commit loss, optional softmax(d).
__global__ __launch_bounds__(256)
void vq_kernel(const __bf16* __restrict__ act, const float* __restrict__ cb,
               const float* __restrict__ en, int K, __bf16* __restrict__ qout,
               float* __restrict__ commit, float* __restrict__ soft,
               int* __restrict__ idxout) {
  const int wave = threadIdx.x >> 5, lane = threadIdx.x & 31;
  const int bt = blockIdx.x * 8 + wave;
  const size_t base = (size_t)bt * F_ + lane * 8;

  float x[8];
  { v8bf xv = *(const v8bf*)(act + base);
    #pragma unroll
    for (int j = 0; j < 8; ++j) x[j] = (float)xv[j]; }

  float ns = 0.f;
  #pragma unroll
  for (int j = 0; j < 8; ++j) ns += x[j] * x[j];
  for (int off = 16; off >= 1; off >>= 1) ns += __shfl_xor(ns, off);
  const float rn = 1.0f / fmaxf(sqrtf(ns), 1e-12f);

  float best = -1e30f; int bi = 0;
  float slot0 = 0.f, slot1 = 0.f;
  for (int k = 0; k < K; ++k) {
    const float* e = en + (size_t)k * F_ + lane * 8;
    float d = 0.f;
    #pragma unroll
    for (int j = 0; j < 8; ++j) d += x[j] * e[j];
    for (int off = 16; off >= 1; off >>= 1) d += __shfl_xor(d, off);
    d *= rn;
    if (d > best) { best = d; bi = k; }
    if (soft) { if (k == lane) slot0 = d; if (k == lane + 32) slot1 = d; }
  }

  const float* q = cb + (size_t)bi * F_ + lane * 8;
  float cs = 0.f;
  v8bf qv;
  #pragma unroll
  for (int j = 0; j < 8; ++j) {
    float qj = q[j];
    qv[j] = (__bf16)qj;
    float df = qj - x[j];
    cs += df * df;
  }
  *(v8bf*)(qout + base) = qv;
  for (int off = 16; off >= 1; off >>= 1) cs += __shfl_xor(cs, off);
  if (lane == 0) {
    atomicAdd(commit, cs * (1.0f / ((float)B_ * F_ * T_)));
    if (idxout) idxout[bt] = bi;
  }
  if (soft) { // K == 48 here
    float m = slot0;
    if (lane < K - 32) m = fmaxf(m, slot1);
    for (int off = 16; off >= 1; off >>= 1) m = fmaxf(m, __shfl_xor(m, off));
    float e0 = __expf(slot0 - m);
    float e1 = (lane < K - 32) ? __expf(slot1 - m) : 0.f;
    float ssum = e0 + e1;
    for (int off = 16; off >= 1; off >>= 1) ssum += __shfl_xor(ssum, off);
    const float inv = 1.0f / ssum;
    soft[(size_t)bt * K + lane] = e0 * inv;
    if (lane < K - 32) soft[(size_t)bt * K + 32 + lane] = e1 * inv;
  }
}

__global__ void smooth_kernel(const int* __restrict__ idx, float* __restrict__ out) {
  __shared__ float red[256];
  const int i = blockIdx.x * 256 + threadIdx.x;
  const int Tm1 = T_ - 1;
  float v = 0.f;
  if (i < B_ * Tm1) {
    int b = i / Tm1, t = i % Tm1;
    v = fabsf((float)(idx[b * T_ + t + 1] - idx[b * T_ + t]));
  }
  red[threadIdx.x] = v;
  __syncthreads();
  for (int s = 128; s > 0; s >>= 1) {
    if (threadIdx.x < s) red[threadIdx.x] += red[threadIdx.x + s];
    __syncthreads();
  }
  if (threadIdx.x == 0) atomicAdd(out, red[0] / ((float)B_ * Tm1));
}

__global__ void init_scalars(float* a, float* b) {
  if (threadIdx.x == 0) { *a = 0.f; *b = 0.f; }
}

extern "C" void kernel_launch(void* const* d_in, const int* in_sizes, int n_in,
                              void* d_out, int out_size, void* d_ws, size_t ws_size,
                              hipStream_t stream) {
  (void)in_sizes; (void)n_in; (void)out_size; (void)ws_size;
  const float* x       = (const float*)d_in[0];
  const float* mask    = (const float*)d_in[1];
  const float* w_in1   = (const float*)d_in[2];
  const float* b_in1   = (const float*)d_in[3];
  const float* s1_dw   = (const float*)d_in[4];
  const float* s1_db   = (const float*)d_in[5];
  const float* s1_pw   = (const float*)d_in[6];
  const float* s1_pb   = (const float*)d_in[7];
  const float* enc_in_w = (const float*)d_in[8];
  const float* enc_in_b = (const float*)d_in[9];
  const float* enc_dw  = (const float*)d_in[10];
  const float* enc_db  = (const float*)d_in[11];
  const float* enc_pw  = (const float*)d_in[12];
  const float* enc_pb  = (const float*)d_in[13];
  const float* dec_in_w = (const float*)d_in[14];
  const float* dec_in_b = (const float*)d_in[15];
  const float* dec_dw  = (const float*)d_in[16];
  const float* dec_db  = (const float*)d_in[17];
  const float* dec_pw  = (const float*)d_in[18];
  const float* dec_pb  = (const float*)d_in[19];
  const float* cb1     = (const float*)d_in[20];
  const float* cb2     = (const float*)d_in[21];
  const float* w_out   = (const float*)d_in[22];
  const float* b_out   = (const float*)d_in[23];

  // ---- workspace carve ----
  char* p = (char*)d_ws;
  auto carve = [&](size_t bytes) { char* r = p; p += (bytes + 255) & ~(size_t)255; return r; };
  __bf16* xb      = (__bf16*)carve((size_t)B_ * T_ * DIM_ * 2);
  __bf16* actA    = (__bf16*)carve((size_t)B_ * T_ * F_ * 2);
  __bf16* actB    = (__bf16*)carve((size_t)B_ * T_ * F_ * 2);
  __bf16* hbuf    = (__bf16*)carve((size_t)B_ * T_ * F_ * 2);
  __bf16* q1      = (__bf16*)carve((size_t)B_ * T_ * F_ * 2);
  __bf16* q2      = (__bf16*)carve((size_t)B_ * T_ * F_ * 2);
  __bf16* wb_in1  = (__bf16*)carve((size_t)F_ * DIM_ * 2);
  __bf16* wb_wout = (__bf16*)carve((size_t)DIM_ * F_ * 2);
  __bf16* wb_s1dw = (__bf16*)carve((size_t)L_ * 3 * F_ * F_ * 2);
  __bf16* wb_s1pw = (__bf16*)carve((size_t)L_ * F_ * F_ * 2);
  __bf16* wb_ein  = (__bf16*)carve((size_t)(S_ - 1) * F_ * F_ * 2);
  __bf16* wb_edw  = (__bf16*)carve((size_t)(S_ - 1) * L_ * 3 * F_ * F_ * 2);
  __bf16* wb_epw  = (__bf16*)carve((size_t)(S_ - 1) * L_ * F_ * F_ * 2);
  __bf16* wb_din  = (__bf16*)carve((size_t)S_ * F_ * F_ * 2);
  __bf16* wb_ddw  = (__bf16*)carve((size_t)S_ * L_ * 3 * F_ * F_ * 2);
  __bf16* wb_dpw  = (__bf16*)carve((size_t)S_ * L_ * F_ * F_ * 2);
  float*  en1     = (float*)carve((size_t)K1_ * F_ * 4);
  float*  en2     = (float*)carve((size_t)K2_ * F_ * 4);
  int*    i2      = (int*)carve((size_t)B_ * T_ * 4);

  // ---- output slices (flat, reference return order) ----
  float* out_main = (float*)d_out;                                   // (B,DIM,T)
  float* out_all  = out_main + (size_t)B_ * DIM_ * T_;               // (S,B,F,T)
  float* commitP  = out_all + (size_t)S_ * B_ * F_ * T_;             // scalar
  float* softP    = commitP + 1;                                     // (B,T,K2)
  float* smoothP  = softP + (size_t)B_ * T_ * K2_;                   // scalar

  const dim3 blk(256), tb(32, 8);

  // ---- precision conversion / layout transforms ----
  transpose_f32_to_bf16<<<dim3(T_ / 32, DIM_ / 32, B_), tb, 0, stream>>>(x, xb, DIM_, T_);
  f32_to_bf16_k<<<1024, 256, 0, stream>>>(w_in1, wb_in1, (size_t)F_ * DIM_);
  f32_to_bf16_k<<<1024, 256, 0, stream>>>(w_out, wb_wout, (size_t)DIM_ * F_);
  f32_to_bf16_k<<<1024, 256, 0, stream>>>(s1_pw, wb_s1pw, (size_t)L_ * F_ * F_);
  f32_to_bf16_k<<<1024, 256, 0, stream>>>(enc_in_w, wb_ein, (size_t)(S_ - 1) * F_ * F_);
  f32_to_bf16_k<<<1024, 256, 0, stream>>>(enc_pw, wb_epw, (size_t)(S_ - 1) * L_ * F_ * F_);
  f32_to_bf16_k<<<1024, 256, 0, stream>>>(dec_in_w, wb_din, (size_t)S_ * F_ * F_);
  f32_to_bf16_k<<<1024, 256, 0, stream>>>(dec_pw, wb_dpw, (size_t)S_ * L_ * F_ * F_);
  conv_w_reorder<<<2048, 256, 0, stream>>>(s1_dw, wb_s1dw, L_);
  conv_w_reorder<<<2048, 256, 0, stream>>>(enc_dw, wb_edw, (S_ - 1) * L_);
  conv_w_reorder<<<2048, 256, 0, stream>>>(dec_dw, wb_ddw, S_ * L_);
  norm_cb<<<K1_, 256, 0, stream>>>(cb1, en1);
  norm_cb<<<K2_, 256, 0, stream>>>(cb2, en2);
  init_scalars<<<1, 32, 0, stream>>>(commitP, smoothP);

  // ---- stage runner: in-GEMM -> 10x (dilated conv + residual pointwise) ----
  auto stage = [&](const __bf16* inAct, int Cin, __bf16* outAct,
                   const __bf16* win, const float* bin,
                   const __bf16* dw, const float* db,
                   const __bf16* pwm, const float* pb) {
    dim3 g(T_ / TN, F_ / TM, B_);
    gemm_pw<false, false, false><<<g, blk, 0, stream>>>(
        win, inAct, bin, nullptr, outAct, nullptr, Cin, F_);
    for (int l = 0; l < L_; ++l) {
      conv_gemm<<<g, blk, 0, stream>>>(dw + (size_t)l * 3 * F_ * F_, outAct,
                                       db + (size_t)l * F_, hbuf, 1 << l);
      gemm_pw<false, true, false><<<g, blk, 0, stream>>>(
          pwm + (size_t)l * F_ * F_, hbuf, pb + (size_t)l * F_,
          outAct, outAct, mask, F_, F_);
    }
  };

  // stage 1 (DIM -> F) + 3 encoder stages; capture output_all slices
  __bf16* cur = actA; __bf16* nxt = actB;
  stage(xb, DIM_, cur, wb_in1, b_in1, wb_s1dw, s1_db, wb_s1pw, s1_pb);
  tc_to_ct_f32<<<dim3(T_ / 32, F_ / 32, B_), tb, 0, stream>>>(cur, out_all, F_, T_);
  for (int s = 0; s < S_ - 1; ++s) {
    stage(cur, F_, nxt,
          wb_ein + (size_t)s * F_ * F_, enc_in_b + (size_t)s * F_,
          wb_edw + (size_t)s * L_ * 3 * F_ * F_, enc_db + (size_t)s * L_ * F_,
          wb_epw + (size_t)s * L_ * F_ * F_, enc_pb + (size_t)s * L_ * F_);
    __bf16* t = cur; cur = nxt; nxt = t;
    tc_to_ct_f32<<<dim3(T_ / 32, F_ / 32, B_), tb, 0, stream>>>(
        cur, out_all + (size_t)(s + 1) * B_ * F_ * T_, F_, T_);
  }

  // ---- double VQ ----
  vq_kernel<<<B_ * T_ / 8, blk, 0, stream>>>(cur, cb1, en1, K1_, q1, commitP, nullptr, nullptr);
  vq_kernel<<<B_ * T_ / 8, blk, 0, stream>>>(q1, cb2, en2, K2_, q2, commitP, softP, i2);
  smooth_kernel<<<(B_ * (T_ - 1) + 255) / 256, blk, 0, stream>>>(i2, smoothP);

  // ---- 4 decoder stages + final projection (F -> DIM, f32, masked) ----
  const __bf16* dcur = q2;
  for (int s = 0; s < S_; ++s) {
    __bf16* dout = (s & 1) ? actB : actA;
    stage(dcur, F_, dout,
          wb_din + (size_t)s * F_ * F_, dec_in_b + (size_t)s * F_,
          wb_ddw + (size_t)s * L_ * 3 * F_ * F_, dec_db + (size_t)s * L_ * F_,
          wb_dpw + (size_t)s * L_ * F_ * F_, dec_pb + (size_t)s * L_ * F_);
    dcur = dout;
  }
  gemm_pw<false, false, true><<<dim3(T_ / TN, DIM_ / TM, B_), blk, 0, stream>>>(
      wb_wout, dcur, b_out, nullptr, out_main, mask, F_, DIM_);
}
